// MupCausalSelfAttention_13821204758686
// MI455X (gfx1250) — matrix operand order repaired and checked by reading, not compiled, not run on hardware
//
#include <hip/hip_runtime.h>

typedef unsigned short ushort_t;
typedef __bf16 v16bf __attribute__((ext_vector_type(16)));
typedef __bf16 v8bf  __attribute__((ext_vector_type(8)));
typedef float  v8f   __attribute__((ext_vector_type(8)));

#define D_MODEL 2048
#define N_HEADS 16
#define D_HEAD  128
#define BATCH   4
#define T_SEQ   2048
#define M_TOT   (BATCH * T_SEQ)   // 8192

static __device__ __forceinline__ unsigned short f2bf(float f) {
    unsigned u = __float_as_uint(f);
    unsigned r = u + 0x7FFFu + ((u >> 16) & 1u);
    return (unsigned short)(r >> 16);
}

static __device__ __forceinline__ v8f vzero8() {
    v8f r;
#pragma unroll
    for (int i = 0; i < 8; ++i) r[i] = 0.0f;
    return r;
}

static __device__ __forceinline__ v16bf concat8(v8bf lo, v8bf hi) {
    v16bf r;
#pragma unroll
    for (int i = 0; i < 8; ++i) { r[i] = lo[i]; r[i + 8] = hi[i]; }
    return r;
}

// gfx1250 async global->LDS copy (16B per lane), tracked by ASYNCcnt.
static __device__ __forceinline__ void async_ld16(unsigned lds_addr,
                                                  const ushort_t* sbase,
                                                  unsigned goff_bytes) {
    asm volatile("global_load_async_to_lds_b128 %0, %1, %2"
                 :: "v"(lds_addr), "v"(goff_bytes), "s"(sbase)
                 : "memory");
}
static __device__ __forceinline__ void wait_async0() {
    asm volatile("s_wait_asynccnt 0x0" ::: "memory");
}

// ---------------------------------------------------------------------------
// 1) fp32 -> bf16 cast (vectorized by 4)
// ---------------------------------------------------------------------------
__global__ __launch_bounds__(256) void cast_bf16_kernel(const float* __restrict__ x,
                                                        ushort_t* __restrict__ y) {
    size_t i = ((size_t)blockIdx.x * 256 + threadIdx.x) * 4;
    float4 f = *(const float4*)(x + i);
    uint2 o;
    o.x = (unsigned)f2bf(f.x) | ((unsigned)f2bf(f.y) << 16);
    o.y = (unsigned)f2bf(f.z) | ((unsigned)f2bf(f.w) << 16);
    *(uint2*)(&y[i]) = o;
}

// ---------------------------------------------------------------------------
// 2) W [K,N] fp32 -> Wt [N,K] bf16 (tiled transpose)
// ---------------------------------------------------------------------------
__global__ __launch_bounds__(256) void transpose_cast_kernel(const float* __restrict__ W,
                                                             ushort_t* __restrict__ Wt,
                                                             int K, int N) {
    __shared__ float tile[32][33];
    int bx = blockIdx.x * 32;  // N
    int by = blockIdx.y * 32;  // K
    int tx = threadIdx.x, ty = threadIdx.y;
#pragma unroll
    for (int i = 0; i < 4; ++i)
        tile[ty + 8 * i][tx] = W[(size_t)(by + ty + 8 * i) * N + (bx + tx)];
    __syncthreads();
#pragma unroll
    for (int i = 0; i < 4; ++i)
        Wt[(size_t)(bx + ty + 8 * i) * K + (by + tx)] = f2bf(tile[tx][ty + 8 * i]);
}

// ---------------------------------------------------------------------------
// 3) GEMM: C[M,N] = A[M,K](bf16) * Bt[N,K]^T(bf16) + bias
//    Double-buffered LDS, tiles filled by async global->LDS (ASYNCcnt).
//    mode 0: scatter to Q[B,H,T,Dh], K[B,H,T,Dh], Vt[B,H,Dh,T] (bf16)
//    mode 1: write fp32 Cout[M,N]
//    Block tile 128x128, 8 waves of 64x32, K-step 32.
// ---------------------------------------------------------------------------
__global__ __launch_bounds__(256) void gemm_bf16_kernel(
    const ushort_t* __restrict__ A, const ushort_t* __restrict__ Bt,
    const float* __restrict__ bias,
    int M, int N, int K, int mode,
    ushort_t* __restrict__ Qh, ushort_t* __restrict__ Kh, ushort_t* __restrict__ Vt,
    float* __restrict__ Cout)
{
    __shared__ alignas(16) ushort_t Abuf[2][128 * 40];
    __shared__ alignas(16) ushort_t Bbuf[2][128 * 40];

    const int tid  = threadIdx.x;
    const int lane = tid & 31;
    const int wave = tid >> 5;
    const int wm   = wave & 1;   // 2 waves along M (64 each)
    const int wn   = wave >> 1;  // 4 waves along N (32 each)
    const int bm   = blockIdx.y * 128;
    const int bn   = blockIdx.x * 128;
    const int l16  = lane & 15;
    const int lh   = lane >> 4;
    const int koff0 = lh * 8;

    v8f acc[4][2];
#pragma unroll
    for (int i = 0; i < 4; ++i)
#pragma unroll
        for (int j = 0; j < 2; ++j) acc[i][j] = vzero8();

    auto issueTile = [&](int kt, int buf) {
#pragma unroll
        for (int i = 0; i < 2; ++i) {
            int idx = tid + i * 256;
            int r = idx >> 2, cc = (idx & 3) * 8;
            unsigned la = (unsigned)(size_t)(&Abuf[buf][r * 40 + cc]);
            unsigned lb = (unsigned)(size_t)(&Bbuf[buf][r * 40 + cc]);
            unsigned ga = (unsigned)(((size_t)(bm + r) * K + kt + cc) * sizeof(ushort_t));
            unsigned gb = (unsigned)(((size_t)(bn + r) * K + kt + cc) * sizeof(ushort_t));
            async_ld16(la, A, ga);
            async_ld16(lb, Bt, gb);
        }
    };

    auto computeTile = [&](int buf) {
        const ushort_t* Asb = Abuf[buf];
        const ushort_t* Bsb = Bbuf[buf];
        v16bf af[4], bfr[2];
#pragma unroll
        for (int fm = 0; fm < 4; ++fm) {
            int row = wm * 64 + fm * 16 + l16;
            const ushort_t* p = &Asb[row * 40 + koff0];
            af[fm] = concat8(*(const v8bf*)p, *(const v8bf*)(p + 16));
        }
#pragma unroll
        for (int fn = 0; fn < 2; ++fn) {
            int row = wn * 32 + fn * 16 + l16;
            const ushort_t* p = &Bsb[row * 40 + koff0];
            bfr[fn] = concat8(*(const v8bf*)p, *(const v8bf*)(p + 16));
        }
#pragma unroll
        for (int fm = 0; fm < 4; ++fm)
#pragma unroll
            for (int fn = 0; fn < 2; ++fn)
                acc[fm][fn] = __builtin_amdgcn_wmma_f32_16x16x32_bf16(
                    false, af[fm], false, bfr[fn], (short)0, acc[fm][fn], false, false);
    };

    // software pipeline: fill buf0, then overlap async fill of buf^1 with WMMAs
    issueTile(0, 0);
    int cur = 0;
    for (int kt = 32; kt < K; kt += 32) {
        wait_async0();       // this wave's fills of buf[cur] have landed
        __syncthreads();     // all waves done filling cur AND done reading cur^1
        issueTile(kt, cur ^ 1);
        computeTile(cur);
        cur ^= 1;
    }
    wait_async0();
    __syncthreads();
    computeTile(cur);

    // epilogue
#pragma unroll
    for (int fn = 0; fn < 2; ++fn) {
        int gn = bn + wn * 32 + fn * 16 + l16;
        float bv = bias[gn];
#pragma unroll
        for (int fm = 0; fm < 4; ++fm) {
            int gmb = bm + wm * 64 + fm * 16 + lh * 8;
            if (mode == 0) {
                int sec = gn >> 11;          // 0=q 1=k 2=v
                int within = gn & 2047;
                int h = within >> 7, d = within & 127;
#pragma unroll
                for (int j = 0; j < 8; ++j) {
                    int gm = gmb + j;
                    int bb = gm >> 11, t = gm & 2047;
                    float v = acc[fm][fn][j] + bv;
                    unsigned short bvv = f2bf(v);
                    if (sec == 0)
                        Qh[((size_t)(bb * N_HEADS + h) * T_SEQ + t) * D_HEAD + d] = bvv;
                    else if (sec == 1)
                        Kh[((size_t)(bb * N_HEADS + h) * T_SEQ + t) * D_HEAD + d] = bvv;
                    else
                        Vt[((size_t)(bb * N_HEADS + h) * D_HEAD + d) * T_SEQ + t] = bvv;
                }
            } else {
#pragma unroll
                for (int j = 0; j < 8; ++j) {
                    int gm = gmb + j;
                    Cout[(size_t)gm * N + gn] = acc[fm][fn][j] + bv;
                }
            }
        }
    }
}

// ---------------------------------------------------------------------------
// 4) Causal flash attention. One wave = 16 query rows.
//    Computes S^T = K*Q^T (cheap softmax reductions), online softmax,
//    O^T += V^T * P^T. Output y in [B,T,H,Dh] bf16.
// ---------------------------------------------------------------------------
__global__ __launch_bounds__(256) void flash_attn_kernel(
    const ushort_t* __restrict__ Qh, const ushort_t* __restrict__ Kh,
    const ushort_t* __restrict__ Vt, ushort_t* __restrict__ Y)
{
    const int tid  = threadIdx.x;
    const int lane = tid & 31;
    const int wave = tid >> 5;
    const int l16  = lane & 15;
    const int lh   = lane >> 4;
    const int koff0 = lh * 8;

    const int bh = blockIdx.y;
    const int bb = bh >> 4, h = bh & 15;
    const int qbase = blockIdx.x * 128 + wave * 16;
    const int myq   = qbase + l16;

    const size_t qkbase = (size_t)bh * T_SEQ * D_HEAD;
    const size_t vbase  = (size_t)bh * D_HEAD * T_SEQ;

    // Preload Q B-fragments: lane = query column, features 8h..8h+7 and 16+8h..16+8h+7
    v16bf qf[4];
#pragma unroll
    for (int c = 0; c < 4; ++c) {
        const ushort_t* p = Qh + qkbase + (size_t)(qbase + l16) * D_HEAD + c * 32 + koff0;
        qf[c] = concat8(*(const v8bf*)p, *(const v8bf*)(p + 16));
    }

    v8f acc[8];
#pragma unroll
    for (int i = 0; i < 8; ++i) acc[i] = vzero8();

    float mcur = -1e30f, lsum = 0.0f;
    const float scale = 1.0f / (float)D_HEAD;
    const int kend = qbase + 16;

    for (int kb = 0; kb < kend; kb += 32) {
        // prefetch next key tile rows (2 cachelines per 256B row, 32 rows/wave)
        int kn = kb + 32;
        if (kn < kend) {
            const char* pk = (const char*)(Kh + qkbase + (size_t)(kn + lane) * D_HEAD);
            __builtin_prefetch(pk, 0, 3);
            __builtin_prefetch(pk + 128, 0, 3);
        }

        // S^T tiles (16 keys x 16 queries each), chained over feature dim
        v8f st[2];
#pragma unroll
        for (int t = 0; t < 2; ++t) {
            int k0 = kb + t * 16;
            const ushort_t* krow = Kh + qkbase + (size_t)(k0 + l16) * D_HEAD;
            v8f s = vzero8();
#pragma unroll
            for (int c = 0; c < 4; ++c) {
                const ushort_t* p = krow + c * 32 + koff0;
                v16bf a = concat8(*(const v8bf*)p, *(const v8bf*)(p + 16));
                s = __builtin_amdgcn_wmma_f32_16x16x32_bf16(
                        false, a, false, qf[c], (short)0, s, false, false);
            }
            st[t] = s;
        }

        // scale + causal mask + running max
        float vals[16];
        float mloc = -1e30f;
#pragma unroll
        for (int t = 0; t < 2; ++t)
#pragma unroll
            for (int j = 0; j < 8; ++j) {
                int key = kb + t * 16 + koff0 + j;
                float sv = st[t][j] * scale;
                if (key > myq) sv = -1e30f;
                vals[t * 8 + j] = sv;
                mloc = fmaxf(mloc, sv);
            }
        mloc = fmaxf(mloc, __shfl_xor(mloc, 16));
        float mnew = fmaxf(mcur, mloc);
        float corr = __expf(mcur - mnew);

        float ls = 0.0f;
        unsigned pk1[4], pk2[4];
#pragma unroll
        for (int i = 0; i < 4; ++i) {
            float p0 = __expf(vals[2 * i]     - mnew);
            float p1 = __expf(vals[2 * i + 1] - mnew);
            float p2 = __expf(vals[8 + 2 * i]     - mnew);
            float p3 = __expf(vals[8 + 2 * i + 1] - mnew);
            ls += p0 + p1 + p2 + p3;
            pk1[i] = (unsigned)f2bf(p0) | ((unsigned)f2bf(p1) << 16);
            pk2[i] = (unsigned)f2bf(p2) | ((unsigned)f2bf(p3) << 16);
        }
        ls += __shfl_xor(ls, 16);
        lsum = lsum * corr + ls;
        mcur = mnew;
#pragma unroll
        for (int dt = 0; dt < 8; ++dt)
#pragma unroll
            for (int j = 0; j < 8; ++j) acc[dt][j] *= corr;

        // Build P B-fragment (lane = query, 16 keys): half of the data comes
        // from the paired lane (xor 16).
        unsigned o1[4], o2[4];
#pragma unroll
        for (int i = 0; i < 4; ++i) {
            o1[i] = __shfl_xor(pk1[i], 16);
            o2[i] = __shfl_xor(pk2[i], 16);
        }
        union { unsigned u[8]; v16bf v; } pu;
#pragma unroll
        for (int i = 0; i < 4; ++i) {
            pu.u[i]     = lh ? o2[i]  : pk1[i];
            pu.u[4 + i] = lh ? pk2[i] : o1[i];
        }
        v16bf pf = pu.v;

        // O^T += V^T * P^T : A-frag rows = feature d, k = 32 keys
#pragma unroll
        for (int dt = 0; dt < 8; ++dt) {
            const ushort_t* p = Vt + vbase + (size_t)(dt * 16 + l16) * T_SEQ + kb + lh * 16;
            v16bf a = concat8(*(const v8bf*)p, *(const v8bf*)(p + 8));
            acc[dt] = __builtin_amdgcn_wmma_f32_16x16x32_bf16(
                          false, a, false, pf, (short)0, acc[dt], false, false);
        }
    }

    float rinv = 1.0f / lsum;
    size_t obase = ((size_t)(bb * T_SEQ + myq) * N_HEADS + h) * D_HEAD;
#pragma unroll
    for (int dt = 0; dt < 8; ++dt)
#pragma unroll
        for (int j = 0; j < 8; ++j) {
            int d = dt * 16 + koff0 + j;
            Y[obase + d] = f2bf(acc[dt][j] * rinv);
        }
}

// ---------------------------------------------------------------------------
// Launch
// ---------------------------------------------------------------------------
extern "C" void kernel_launch(void* const* d_in, const int* in_sizes, int n_in,
                              void* d_out, int out_size, void* d_ws, size_t ws_size,
                              hipStream_t stream) {
    (void)in_sizes; (void)n_in; (void)out_size; (void)ws_size;
    const float* x     = (const float*)d_in[0];
    const float* W_qkv = (const float*)d_in[1];
    const float* b_qkv = (const float*)d_in[2];
    const float* W_out = (const float*)d_in[3];
    const float* b_out = (const float*)d_in[4];
    float* out = (float*)d_out;

    char* ws = (char*)d_ws;
    size_t o = 0;
    ushort_t* xb     = (ushort_t*)(ws + o); o += (size_t)M_TOT * D_MODEL * 2;         // 32MB
    ushort_t* wqkvt  = (ushort_t*)(ws + o); o += (size_t)(3 * D_MODEL) * D_MODEL * 2; // 24MB
    ushort_t* woutt  = (ushort_t*)(ws + o); o += (size_t)D_MODEL * D_MODEL * 2;       // 8MB
    ushort_t* Qh     = (ushort_t*)(ws + o); o += (size_t)BATCH * N_HEADS * T_SEQ * D_HEAD * 2;
    ushort_t* Kh     = (ushort_t*)(ws + o); o += (size_t)BATCH * N_HEADS * T_SEQ * D_HEAD * 2;
    ushort_t* Vt     = (ushort_t*)(ws + o); o += (size_t)BATCH * N_HEADS * D_HEAD * T_SEQ * 2;
    ushort_t* yb     = (ushort_t*)(ws + o); o += (size_t)M_TOT * D_MODEL * 2;

    // x -> bf16
    cast_bf16_kernel<<<(M_TOT * D_MODEL) / 1024, 256, 0, stream>>>(x, xb);
    // W_qkv [2048,6144] -> [6144,2048] bf16 ; W_out -> transposed bf16
    transpose_cast_kernel<<<dim3(3 * D_MODEL / 32, D_MODEL / 32), dim3(32, 8), 0, stream>>>(
        W_qkv, wqkvt, D_MODEL, 3 * D_MODEL);
    transpose_cast_kernel<<<dim3(D_MODEL / 32, D_MODEL / 32), dim3(32, 8), 0, stream>>>(
        W_out, woutt, D_MODEL, D_MODEL);

    // QKV projection -> scattered Q/K/V(transposed)
    gemm_bf16_kernel<<<dim3(3 * D_MODEL / 128, M_TOT / 128), 256, 0, stream>>>(
        xb, wqkvt, b_qkv, M_TOT, 3 * D_MODEL, D_MODEL, 0, Qh, Kh, Vt, nullptr);

    // causal flash attention
    flash_attn_kernel<<<dim3(T_SEQ / 128, BATCH * N_HEADS), 256, 0, stream>>>(Qh, Kh, Vt, yb);

    // output projection -> fp32 out
    gemm_bf16_kernel<<<dim3(D_MODEL / 128, M_TOT / 128), 256, 0, stream>>>(
        yb, woutt, b_out, M_TOT, D_MODEL, D_MODEL, 1, nullptr, nullptr, nullptr, out);
}
